// MoTGPT2Block_62448824483993
// MI455X (gfx1250) — matrix operand order repaired and checked by reading, not compiled, not run on hardware
//
#include <hip/hip_runtime.h>

// ---------------------------------------------------------------------------
// MoT GPT-2 block for MI455X (gfx1250, wave32, WMMA, bf16 compute / f32 accum)
// k0a: f32 -> bf16 convert hidden_states, pre-masked per modality
// k0b: f32 -> bf16 convert + transpose (W_attn, W_proj -> [n][k])
// k1 : qkv = sum_m( x_m_masked @ W_attn[m] ) + b_attn[tid]; Q pre-scaled by
//      0.125*log2(e) so attention softmax runs in exp2 domain
// k1b: transpose V section of qkv -> VT[b,h,hd,s]
// k2 : causal flash attention per (b, h, 64-row q tile)  (bf16 WMMA)
// k3 : out[m] = mask_m * (attn @ W_proj[m] + b_proj[m])  (bf16 WMMA GEMM)
// Tile fills: GLOBAL_LOAD_ASYNC_TO_LDS_B64, double-buffered and software-
// pipelined with partial s_wait_asynccnt; fragments load as 16B-aligned
// ds_load_b128 pairs.
// ---------------------------------------------------------------------------

typedef __attribute__((ext_vector_type(16))) __bf16        v16bf;
typedef __attribute__((ext_vector_type(8)))  float         v8f;
typedef __attribute__((ext_vector_type(4)))  float         v4f;
typedef __attribute__((ext_vector_type(4)))  unsigned int  v4u;
typedef __attribute__((ext_vector_type(2)))  unsigned int  v2u;
typedef int v2i_a __attribute__((vector_size(8)));   // builtin's b64 data type

#define MODS   2
#define BATCH  4
#define SEQ    1024
#define DIM    1024
#define HEADS  16
#define HD     64
#define N3     3072
#define ROWS   4096   // B*S

#define QSCALE 0.18033688011112042f   // (1/sqrt(64)) * log2(e)

#ifndef __has_builtin
#define __has_builtin(x) 0
#endif
#if __has_builtin(__builtin_amdgcn_global_load_async_to_lds_b64) && \
    __has_builtin(__builtin_amdgcn_s_wait_asynccnt)
#define HAVE_ASYNC_LDS 1
#define WAIT_ASYNC(n) __builtin_amdgcn_s_wait_asynccnt(n)
#else
#define HAVE_ASYNC_LDS 0
#define WAIT_ASYNC(n)
#endif

__device__ __forceinline__ unsigned f2bfu(float f) {
    unsigned u = __float_as_uint(f);
    u += 0x7FFFu + ((u >> 16) & 1u);   // round to nearest even
    return u >> 16;
}

__device__ __forceinline__ v8f wmma_bf16(v16bf a, v16bf b, v8f c) {
    return __builtin_amdgcn_wmma_f32_16x16x32_bf16(
        false, a, false, b, (short)0, c, false, false);
}

// 8-byte tile-fill copy: async global->LDS when available, else through VGPRs.
__device__ __forceinline__ void copy_b64(unsigned short* lds,
                                         const unsigned short* g) {
#if HAVE_ASYNC_LDS
    __builtin_amdgcn_global_load_async_to_lds_b64(
        (__attribute__((address_space(1))) v2i_a*)(g),
        (__attribute__((address_space(3))) v2i_a*)(lds), 0, 0);
#else
    *(v2u*)lds = *(const v2u*)g;
#endif
}

// A fragment from a 32-element bf16 k-line (row m of A tile).
// frag dword r <-> k-dword (r<4?0:8)+half*4+(r&3): chunks [half], [2+half].
__device__ __forceinline__ v16bf load_a_line(const unsigned short* line, int half) {
    const v4u* p = (const v4u*)line;
    union { v4u u[2]; v16bf b; } u;
    u.u[0] = p[half];
    u.u[1] = p[2 + half];
    return u.b;
}

// B fragment from a 32-element bf16 k-line (col n of B, stored transposed).
// frag dword r <-> k-dword (r<4?0:8)+half*4+(r&3): chunks [half], [2+half].
__device__ __forceinline__ v16bf load_b_line(const unsigned short* line, int half) {
    const v4u* p = (const v4u*)line;
    union { v4u u[2]; v16bf b; } u;
    u.u[0] = p[half];
    u.u[1] = p[2 + half];
    return u.b;
}

// 4x4 bf16 block transpose (bf16 src row-major -> bf16 dst row-major swapped).
__device__ __forceinline__ void transpose4x4_bf(
    const unsigned short* __restrict__ src, size_t srcStride,
    unsigned short* dst, int dstStride, int r0, int c0)
{
    v2u a0 = *(const v2u*)(src + (size_t)(r0 + 0) * srcStride + c0);
    v2u a1 = *(const v2u*)(src + (size_t)(r0 + 1) * srcStride + c0);
    v2u a2 = *(const v2u*)(src + (size_t)(r0 + 2) * srcStride + c0);
    v2u a3 = *(const v2u*)(src + (size_t)(r0 + 3) * srcStride + c0);
    v2u t0, t1, t2, t3;
    t0.x = (a0.x & 0xffffu) | (a1.x << 16);     t0.y = (a2.x & 0xffffu) | (a3.x << 16);
    t1.x = (a0.x >> 16) | (a1.x & 0xffff0000u); t1.y = (a2.x >> 16) | (a3.x & 0xffff0000u);
    t2.x = (a0.y & 0xffffu) | (a1.y << 16);     t2.y = (a2.y & 0xffffu) | (a3.y << 16);
    t3.x = (a0.y >> 16) | (a1.y & 0xffff0000u); t3.y = (a2.y >> 16) | (a3.y & 0xffff0000u);
    *(v2u*)(dst + (size_t)(c0 + 0) * dstStride + r0) = t0;
    *(v2u*)(dst + (size_t)(c0 + 1) * dstStride + r0) = t1;
    *(v2u*)(dst + (size_t)(c0 + 2) * dstStride + r0) = t2;
    *(v2u*)(dst + (size_t)(c0 + 3) * dstStride + r0) = t3;
}

// ---------------------------------------------------------------------------
// Kernel 0a: hidden_states f32 -> bf16, pre-masked by modality:
//            dst[m,row,:] = (type_ids[row]==m) ? bf16(src[m,row,:]) : 0
// ---------------------------------------------------------------------------
__global__ __launch_bounds__(256) void cvt_hs_kernel(
    const float* __restrict__ src, const int* __restrict__ tid,
    unsigned short* __restrict__ dst)
{
    int i = blockIdx.x * 256 + threadIdx.x;   // one 4-element group
    size_t base = (size_t)i * 4;
    int m   = (int)(base >> 22);              // / (ROWS*DIM)
    int row = (int)((base >> 10) & (ROWS - 1));
    v4f v = *(const v4f*)(src + base);
    v2u o;
    if (tid[row] == m) {
        o.x = (f2bfu(v.y) << 16) | f2bfu(v.x);
        o.y = (f2bfu(v.w) << 16) | f2bfu(v.z);
    } else {
        o.x = 0u; o.y = 0u;
    }
    *(v2u*)(dst + base) = o;
}

// ---------------------------------------------------------------------------
// Kernel 0b: f32 [K,N] -> bf16 [N,K] (convert + transpose, 4x4 / thread)
// ---------------------------------------------------------------------------
__global__ __launch_bounds__(256) void cvtT_bf16_kernel(
    const float* __restrict__ src, unsigned short* __restrict__ dst,
    int K, int N)
{
    int i = blockIdx.x * 256 + threadIdx.x;
    int n4 = N >> 2;
    int kb = (i / n4) * 4;
    int nb = (i % n4) * 4;
    if (kb >= K) return;
    v4f r0 = *(const v4f*)(src + (size_t)(kb + 0) * N + nb);
    v4f r1 = *(const v4f*)(src + (size_t)(kb + 1) * N + nb);
    v4f r2 = *(const v4f*)(src + (size_t)(kb + 2) * N + nb);
    v4f r3 = *(const v4f*)(src + (size_t)(kb + 3) * N + nb);
    v2u o;
    o.x = (f2bfu(r1.x) << 16) | f2bfu(r0.x);
    o.y = (f2bfu(r3.x) << 16) | f2bfu(r2.x);
    *(v2u*)(dst + (size_t)(nb + 0) * K + kb) = o;
    o.x = (f2bfu(r1.y) << 16) | f2bfu(r0.y);
    o.y = (f2bfu(r3.y) << 16) | f2bfu(r2.y);
    *(v2u*)(dst + (size_t)(nb + 1) * K + kb) = o;
    o.x = (f2bfu(r1.z) << 16) | f2bfu(r0.z);
    o.y = (f2bfu(r3.z) << 16) | f2bfu(r2.z);
    *(v2u*)(dst + (size_t)(nb + 2) * K + kb) = o;
    o.x = (f2bfu(r1.w) << 16) | f2bfu(r0.w);
    o.y = (f2bfu(r3.w) << 16) | f2bfu(r2.w);
    *(v2u*)(dst + (size_t)(nb + 3) * K + kb) = o;
}

// ---------------------------------------------------------------------------
// Kernel 1: QKV projection. 128 thr (4 waves), 128x64 tile of [4096, 3072].
// 64 pipelined k-steps (2 modalities x 32); double-buffered async LDS tiles.
// Each wave: 32 rows x 64 cols = 8 WMMA per step.
// ---------------------------------------------------------------------------
#define QKV_AT (128 * 40)
#define QKV_BT (64 * 40)

__global__ __launch_bounds__(128) void qkv_kernel(
    const unsigned short* __restrict__ hsb,  // bf16 [M, 4096, 1024] pre-masked
    const unsigned short* __restrict__ WaT,  // bf16 [M, 3072, 1024] transposed
    const float* __restrict__ ba,            // [M, 3072]
    const int*   __restrict__ tid,           // [4096]
    unsigned short* __restrict__ qkv)        // bf16 [4096, 3072]
{
    __shared__ __align__(16) unsigned short As[2 * QKV_AT];
    __shared__ __align__(16) unsigned short BsT[2 * QKV_BT];

    const int lane = threadIdx.x & 31;
    const int wave = threadIdx.x >> 5;
    const int half = lane >> 4;
    const int ln   = lane & 15;
    const int rowBase = blockIdx.y * 128;
    const int colBase = blockIdx.x * 64;

    v8f acc[2][4];
#pragma unroll
    for (int at = 0; at < 2; ++at)
#pragma unroll
        for (int nt = 0; nt < 4; ++nt)
#pragma unroll
            for (int i = 0; i < 8; ++i) acc[at][nt][i] = 0.0f;

    // issue async copies for step kk into buffer buf (12 async ops / wave)
    auto issue = [&](int kk, int buf) {
        const int mm = kk >> 5;
        const int k0 = (kk & 31) * 32;
        const unsigned short* hsm = hsb + (size_t)mm * ROWS * DIM;
        const unsigned short* WmT = WaT + (size_t)mm * N3 * DIM;
        unsigned short* Ad = As + buf * QKV_AT;
        unsigned short* Bd = BsT + buf * QKV_BT;
#pragma unroll
        for (int it = 0; it < 8; ++it) {
            int e = threadIdx.x + it * 128;
            int ar = e >> 3, ak = (e & 7) * 4;
            copy_b64(Ad + ar * 40 + ak,
                     hsm + (size_t)(rowBase + ar) * DIM + k0 + ak);
        }
#pragma unroll
        for (int it = 0; it < 4; ++it) {
            int e = threadIdx.x + it * 128;
            int n = e >> 3, nk = (e & 7) * 4;
            copy_b64(Bd + n * 40 + nk,
                     WmT + (size_t)(colBase + n) * DIM + k0 + nk);
        }
        __builtin_prefetch(WmT + (size_t)colBase * DIM + k0 + 64, 0, 0);
    };
    auto compute = [&](int buf) {
        const unsigned short* Ab = As + buf * QKV_AT;
        const unsigned short* Bb = BsT + buf * QKV_BT;
        v16bf a0 = load_a_line(Ab + (wave * 32 + ln) * 40, half);
        v16bf a1 = load_a_line(Ab + (wave * 32 + 16 + ln) * 40, half);
#pragma unroll
        for (int nt = 0; nt < 4; ++nt) {
            v16bf b = load_b_line(Bb + (nt * 16 + ln) * 40, half);
            acc[0][nt] = wmma_bf16(a0, b, acc[0][nt]);
            acc[1][nt] = wmma_bf16(a1, b, acc[1][nt]);
        }
    };

    issue(0, 0);
    for (int kk = 0; kk < 63; ++kk) {
        issue(kk + 1, (kk + 1) & 1);
        WAIT_ASYNC(12);          // previous tile's copies complete (in-order)
        __syncthreads();
        compute(kk & 1);
        __syncthreads();         // readers done before buffer reuse
    }
    WAIT_ASYNC(0);
    __syncthreads();
    compute(1);                  // kk = 63

    // Q section (cols < DIM) pre-scaled for exp2-domain softmax
    const float oscale = (colBase < DIM) ? QSCALE : 1.0f;
#pragma unroll
    for (int at = 0; at < 2; ++at)
#pragma unroll
        for (int nt = 0; nt < 4; ++nt)
#pragma unroll
            for (int i = 0; i < 8; ++i) {
                int row = rowBase + wave * 32 + at * 16 + half * 8 + i;
                int col = colBase + nt * 16 + ln;
                float v = (acc[at][nt][i] + ba[(size_t)tid[row] * N3 + col]) * oscale;
                qkv[(size_t)row * N3 + col] = (unsigned short)f2bfu(v);
            }
}

// ---------------------------------------------------------------------------
// Kernel 1b: transpose V section of qkv -> VT[b*16+h][hd][s]  (one-shot)
// ---------------------------------------------------------------------------
__global__ __launch_bounds__(256) void vt_kernel(
    const unsigned short* __restrict__ qkv,  // bf16 [4096, 3072]
    unsigned short* __restrict__ VT)         // bf16 [64, 64, 1024]
{
    int t = blockIdx.x * 256 + threadIdx.x;  // 262144 4x4 blocks
    int d4 = (t & 15) * 4;
    int s4 = ((t >> 4) & 255) * 4;
    int bh = t >> 12;                        // 0..63
    int b = bh >> 4, h = bh & 15;
    const unsigned short* src = qkv + (size_t)b * SEQ * N3 + 2 * DIM + h * HD;
    unsigned short* dst = VT + (size_t)bh * HD * SEQ;
    transpose4x4_bf(src, N3, dst, SEQ, s4, d4);
}

// ---------------------------------------------------------------------------
// Kernel 2: causal flash attention (exp2 domain). 128 thr, one (b, h, q tile).
// K/V tiles double-buffered and pipelined across the kt loop.
// ---------------------------------------------------------------------------
#define ATT_KT (64 * 72)

__global__ __launch_bounds__(128) void attn_kernel(
    const unsigned short* __restrict__ qkv,  // bf16 [4096, 3072] (Q pre-scaled)
    const unsigned short* __restrict__ VT,   // bf16 [64, 64, 1024]
    unsigned short* __restrict__ attn)       // bf16 [4096, 1024]
{
    __shared__ __align__(16) unsigned short Ks[2 * ATT_KT];   // [krow][hd]
    __shared__ __align__(16) unsigned short Vs[2 * ATT_KT];   // [hd][krow]
    __shared__ __align__(16) unsigned short Ps[4 * 16 * 40];  // per-wave 16x64

    const int lane = threadIdx.x & 31;
    const int wave = threadIdx.x >> 5;
    const int half = lane >> 4;
    const int ln   = lane & 15;

    const int qt = blockIdx.x;
    const int h  = blockIdx.y;
    const int b  = blockIdx.z;
    const int qBase = qt * 64;
    const size_t rowb = (size_t)b * SEQ;
    const unsigned short* vsrc = VT + (size_t)(b * HEADS + h) * HD * SEQ;

    // Q fragments: two b128 global loads per frag (one row per lane)
    const unsigned short* qrow =
        qkv + (rowb + qBase + wave * 16 + ln) * N3 + h * HD;
    v16bf qf0 = load_a_line(qrow,      half);
    v16bf qf1 = load_a_line(qrow + 32, half);

    float mi[8], li[8];
    v8f o[4];
#pragma unroll
    for (int i = 0; i < 8; ++i) { mi[i] = -1e30f; li[i] = 0.0f; }
#pragma unroll
    for (int nt = 0; nt < 4; ++nt)
#pragma unroll
        for (int i = 0; i < 8; ++i) o[nt][i] = 0.0f;

    // issue async K/V tile copies for step kt into buffer buf (16 ops / wave)
    auto issueKV = [&](int kt, int buf) {
        const int kBase = kt * 64;
        const unsigned short* ksrc = qkv + (rowb + kBase) * N3 + DIM + h * HD;
        unsigned short* Kd = Ks + buf * ATT_KT;
        unsigned short* Vd = Vs + buf * ATT_KT;
#pragma unroll
        for (int it = 0; it < 8; ++it) {
            int e = threadIdx.x + it * 128;
            int r = e >> 4, c = (e & 15) * 4;
            copy_b64(Kd + r * 72 + c, ksrc + (size_t)r * N3 + c);
            copy_b64(Vd + r * 72 + c, vsrc + (size_t)r * SEQ + kBase + c);
        }
    };

    issueKV(0, 0);
    for (int kt = 0; kt <= qt; ++kt) {
        const int buf = kt & 1;
        const int kBase = kt * 64;
        if (kt < qt) {
            issueKV(kt + 1, buf ^ 1);
            WAIT_ASYNC(16);
        } else {
            WAIT_ASYNC(0);
        }
        __syncthreads();

        // scores (log2-domain): Q(16x64) @ K^T -> 4 tiles of 16x16
        const unsigned short* Kb = Ks + buf * ATT_KT;
        const unsigned short* Vb = Vs + buf * ATT_KT;
        v8f st[4];
#pragma unroll
        for (int nt = 0; nt < 4; ++nt) {
#pragma unroll
            for (int i = 0; i < 8; ++i) st[nt][i] = 0.0f;
            const unsigned short* kline = Kb + (nt * 16 + ln) * 72;
            st[nt] = wmma_bf16(qf0, load_b_line(kline,      half), st[nt]);
            st[nt] = wmma_bf16(qf1, load_b_line(kline + 32, half), st[nt]);
        }

        // online softmax via exp2; row = 8*half + i within wave's 16 q rows
        unsigned short* pw = Ps + wave * 16 * 40;
#pragma unroll
        for (int i = 0; i < 8; ++i) {
            int qg = qBase + wave * 16 + half * 8 + i;
            float sv[4];
            float rmax = -1e30f;
#pragma unroll
            for (int nt = 0; nt < 4; ++nt) {
                int kg = kBase + nt * 16 + ln;
                float s = st[nt][i];
                if (kg > qg) s = -1e30f;           // causal mask
                sv[nt] = s;
                rmax = fmaxf(rmax, s);
            }
#pragma unroll
            for (int off = 1; off < 16; off <<= 1)
                rmax = fmaxf(rmax, __shfl_xor(rmax, off, 32));
            float mnew  = fmaxf(mi[i], rmax);
            float alpha = exp2f(mi[i] - mnew);
            float rsum = 0.0f;
#pragma unroll
            for (int nt = 0; nt < 4; ++nt) {
                float p = exp2f(sv[nt] - mnew);
                rsum += p;
                pw[(half * 8 + i) * 40 + nt * 16 + ln] = (unsigned short)f2bfu(p);
            }
#pragma unroll
            for (int off = 1; off < 16; off <<= 1)
                rsum += __shfl_xor(rsum, off, 32);
            li[i] = li[i] * alpha + rsum;
            mi[i] = mnew;
#pragma unroll
            for (int nt = 0; nt < 4; ++nt) o[nt][i] *= alpha;
        }

        // P(16x64) @ V: P as A frags (per-wave LDS), V B-frags from Vb
        const unsigned short* pline = pw + ln * 40;
        v16bf pf0 = load_a_line(pline,      half);
        v16bf pf1 = load_a_line(pline + 32, half);
#pragma unroll
        for (int nt = 0; nt < 4; ++nt) {
            const unsigned short* vline = Vb + (nt * 16 + ln) * 72;
            o[nt] = wmma_bf16(pf0, load_b_line(vline,      half), o[nt]);
            o[nt] = wmma_bf16(pf1, load_b_line(vline + 32, half), o[nt]);
        }
        __syncthreads();
    }

    // normalize and store bf16 attn output [4096, 1024]
#pragma unroll
    for (int i = 0; i < 8; ++i) {
        float inv = 1.0f / li[i];
        int srow = qBase + wave * 16 + half * 8 + i;
#pragma unroll
        for (int nt = 0; nt < 4; ++nt) {
            int col = h * HD + nt * 16 + ln;
            attn[(rowb + srow) * (size_t)DIM + col] =
                (unsigned short)f2bfu(o[nt][i] * inv);
        }
    }
}

// ---------------------------------------------------------------------------
// Kernel 3: output projection + masked scatter. 128x64 tiles; grid z = m.
// 32 pipelined k-steps, double-buffered async LDS tiles.
// ---------------------------------------------------------------------------
__global__ __launch_bounds__(128) void proj_kernel(
    const unsigned short* __restrict__ attn, // bf16 [4096, 1024]
    const unsigned short* __restrict__ WpT,  // bf16 [M, 1024, 1024] transposed
    const float* __restrict__ bp,            // [M, 1024]
    const int*   __restrict__ tid,           // [4096]
    float* __restrict__ out)                 // [M, 4096, 1024]
{
    __shared__ __align__(16) unsigned short As[2 * QKV_AT];
    __shared__ __align__(16) unsigned short BsT[2 * QKV_BT];

    const int lane = threadIdx.x & 31;
    const int wave = threadIdx.x >> 5;
    const int half = lane >> 4;
    const int ln   = lane & 15;
    const int rowBase = blockIdx.y * 128;
    const int colBase = blockIdx.x * 64;
    const int m = blockIdx.z;
    const unsigned short* WmT = WpT + (size_t)m * DIM * DIM;

    v8f acc[2][4];
#pragma unroll
    for (int at = 0; at < 2; ++at)
#pragma unroll
        for (int nt = 0; nt < 4; ++nt)
#pragma unroll
            for (int i = 0; i < 8; ++i) acc[at][nt][i] = 0.0f;

    auto issue = [&](int ks, int buf) {
        const int k0 = ks * 32;
        unsigned short* Ad = As + buf * QKV_AT;
        unsigned short* Bd = BsT + buf * QKV_BT;
#pragma unroll
        for (int it = 0; it < 8; ++it) {
            int e = threadIdx.x + it * 128;
            int ar = e >> 3, ak = (e & 7) * 4;
            copy_b64(Ad + ar * 40 + ak,
                     attn + (size_t)(rowBase + ar) * DIM + k0 + ak);
        }
#pragma unroll
        for (int it = 0; it < 4; ++it) {
            int e = threadIdx.x + it * 128;
            int n = e >> 3, nk = (e & 7) * 4;
            copy_b64(Bd + n * 40 + nk,
                     WmT + (size_t)(colBase + n) * DIM + k0 + nk);
        }
        __builtin_prefetch(WmT + (size_t)colBase * DIM + k0 + 64, 0, 0);
    };
    auto compute = [&](int buf) {
        const unsigned short* Ab = As + buf * QKV_AT;
        const unsigned short* Bb = BsT + buf * QKV_BT;
        v16bf a0 = load_a_line(Ab + (wave * 32 + ln) * 40, half);
        v16bf a1 = load_a_line(Ab + (wave * 32 + 16 + ln) * 40, half);
#pragma unroll
        for (int nt = 0; nt < 4; ++nt) {
            v16bf b = load_b_line(Bb + (nt * 16 + ln) * 40, half);
            acc[0][nt] = wmma_bf16(a0, b, acc[0][nt]);
            acc[1][nt] = wmma_bf16(a1, b, acc[1][nt]);
        }
    };

    issue(0, 0);
    for (int ks = 0; ks < 31; ++ks) {
        issue(ks + 1, (ks + 1) & 1);
        WAIT_ASYNC(12);
        __syncthreads();
        compute(ks & 1);
        __syncthreads();
    }
    WAIT_ASYNC(0);
    __syncthreads();
    compute(1);                  // ks = 31

#pragma unroll
    for (int at = 0; at < 2; ++at)
#pragma unroll
        for (int nt = 0; nt < 4; ++nt)
#pragma unroll
            for (int i = 0; i < 8; ++i) {
                int row = rowBase + wave * 32 + at * 16 + half * 8 + i;
                int col = colBase + nt * 16 + ln;
                float v = (tid[row] == m)
                            ? (acc[at][nt][i] + bp[(size_t)m * DIM + col]) : 0.0f;
                out[((size_t)m * ROWS + row) * DIM + col] = v;
            }
}

// ---------------------------------------------------------------------------
extern "C" void kernel_launch(void* const* d_in, const int* in_sizes, int n_in,
                              void* d_out, int out_size, void* d_ws, size_t ws_size,
                              hipStream_t stream) {
    const float* hs = (const float*)d_in[0];   // [2,4,1024,1024]
    const float* Wa = (const float*)d_in[1];   // [2,1024,3072]
    const float* ba = (const float*)d_in[2];   // [2,3072]
    const float* Wp = (const float*)d_in[3];   // [2,1024,1024]
    const float* bp = (const float*)d_in[4];   // [2,1024]
    const int*   ti = (const int*)d_in[5];     // [4096]
    float* out = (float*)d_out;                // [2,4,1024,1024] f32

    // workspace layout (ushort elements), 72 MB total
    unsigned short* qkv_ws  = (unsigned short*)d_ws;                 // 12582912
    unsigned short* attn_ws = qkv_ws  + (size_t)ROWS * N3;           //  4194304
    unsigned short* hs_bf   = attn_ws + (size_t)ROWS * DIM;          //  8388608
    unsigned short* WaT_bf  = hs_bf   + (size_t)MODS * ROWS * DIM;   //  6291456
    unsigned short* WpT_bf  = WaT_bf  + (size_t)MODS * DIM * N3;     //  2097152
    unsigned short* VT_ws   = WpT_bf  + (size_t)MODS * DIM * DIM;    //  4194304

    const int n4_hs = MODS * ROWS * DIM / 4;                 // 2097152
    cvt_hs_kernel<<<n4_hs / 256, 256, 0, stream>>>(hs, ti, hs_bf);
    for (int m = 0; m < MODS; ++m) {
        cvtT_bf16_kernel<<<(DIM / 4) * (N3 / 4) / 256, 256, 0, stream>>>(
            Wa + (size_t)m * DIM * N3, WaT_bf + (size_t)m * N3 * DIM, DIM, N3);
        cvtT_bf16_kernel<<<(DIM / 4) * (DIM / 4) / 256, 256, 0, stream>>>(
            Wp + (size_t)m * DIM * DIM, WpT_bf + (size_t)m * DIM * DIM, DIM, DIM);
    }

    qkv_kernel <<<dim3(N3 / 64, ROWS / 128),        128, 0, stream>>>(hs_bf, WaT_bf, ba, ti, qkv_ws);
    vt_kernel  <<<BATCH * HEADS * (SEQ/4) * (HD/4) / 256, 256, 0, stream>>>(qkv_ws, VT_ws);
    attn_kernel<<<dim3(SEQ / 64, HEADS, BATCH),     128, 0, stream>>>(qkv_ws, VT_ws, attn_ws);
    proj_kernel<<<dim3(DIM / 64, ROWS / 128, MODS), 128, 0, stream>>>(attn_ws, WpT_bf, bp, ti, out);
}
